// mamba_ref_35055523070795
// MI455X (gfx1250) — compile-verified, matrix operand order checked by reading, not run on hardware
//
#include <hip/hip_runtime.h>
#include <hip/hip_bf16.h>

// ---------------------------------------------------------------------------
// Mamba block forward for MI455X (gfx1250, wave32, WMMA).
// GEMMs: bf16 operands (one-shot f32->bf16 pre-convert), double-buffered LDS
//        fed by GLOBAL_LOAD_ASYNC_TO_LDS_B128 (ASYNCcnt), 64x64 tile per wave,
//        32 WMMA per K-stage (BK=64).
// Scan:  1 channel per lane, 16 states in VGPRs, B/C staged via LDS.
// ---------------------------------------------------------------------------

typedef __attribute__((ext_vector_type(16))) __bf16 v16bf;
typedef __attribute__((ext_vector_type(8)))  __bf16 v8bf;
typedef __attribute__((ext_vector_type(8)))  float  v8f;

#define BM 128
#define BN 256
#define BK 64   // two WMMA k-steps per LDS stage

// --- gfx1250 async global->LDS copy (GVS mode), tracked by ASYNCcnt ---------
__device__ __forceinline__ void async_ld_b128(unsigned lds_addr, const void* base,
                                              int voff_bytes) {
    asm volatile("global_load_async_to_lds_b128 %0, %1, %2"
                 :: "v"(lds_addr), "v"(voff_bytes), "s"(base) : "memory");
}
__device__ __forceinline__ void wait_async0() {
    asm volatile("s_wait_asynccnt 0x0" ::: "memory");
}
// LDS byte address = low 32 bits of the flat address (LDS aperture keeps the
// wave-relative offset in addr[31:0]).
__device__ __forceinline__ unsigned lds_addr_of(const void* p) {
    return (unsigned)(unsigned long long)p;
}

// C[M,N] = A[M,K] * W[N,K]^T  (A, W bf16 row-major, K contiguous)
// mode 0: plain f32 store.  mode 1: softplus(acc + bias[n]).
// Cbf != nullptr: additionally store bf16 copy of C (same ldc).
__global__ __launch_bounds__(256) void gemm_bf16_async(
    const __bf16* __restrict__ A, int lda,
    const __bf16* __restrict__ W, int ldw,
    float* __restrict__ C, int ldc, __bf16* __restrict__ Cbf,
    int M, int N, int K,
    int mode, const float* __restrict__ bias)
{
    __shared__ __bf16 As[2][BM * BK];   // 2 x 16 KB
    __shared__ __bf16 Ws[2][BN * BK];   // 2 x 32 KB

    const int tid  = threadIdx.x;
    const int lane = tid & 31;
    const int wid  = tid >> 5;
    const int wm   = wid >> 2;        // 0..1  -> 64 rows each
    const int wn   = wid & 3;         // 0..3  -> 64 cols each
    const int m_block = blockIdx.y * BM;
    const int n_block = blockIdx.x * BN;
    const int half = lane >> 4;       // K-phase per ISA 16-bit frag layout
    const int l15  = lane & 15;

    // ---- staging geometry: b128 = 8 bf16; row = srow + 32*j, col = scol ----
    const int srow = tid >> 3;            // 0..31
    const int scol = (tid & 7) << 3;      // 0..56 (bf16 elements)

    int  aoff[4];
#pragma unroll
    for (int j = 0; j < 4; ++j)
        aoff[j] = (int)(((size_t)(m_block + srow + 32 * j) * lda + scol) * 2);

    int  woff[8];
    bool wvalid[8];
#pragma unroll
    for (int j = 0; j < 8; ++j) {
        int nr  = n_block + srow + 32 * j;
        wvalid[j] = (nr < N);
        int nc  = wvalid[j] ? nr : 0;
        woff[j] = (int)(((size_t)nc * ldw + scol) * 2);
    }
    const unsigned aL = lds_addr_of(&As[0][srow * BK + scol]);
    const unsigned wL = lds_addr_of(&Ws[0][srow * BK + scol]);
    const unsigned ABUF = BM * BK * 2;    // bytes per A buffer
    const unsigned WBUF = BN * BK * 2;
    const unsigned JSTRIDE = 32 * BK * 2; // bytes per j step

    // pre-zero LDS rows for invalid W rows (their lanes never issue asyncs)
#pragma unroll
    for (int j = 0; j < 8; ++j) {
        if (!wvalid[j]) {
            v8bf z = {(__bf16)0.f,(__bf16)0.f,(__bf16)0.f,(__bf16)0.f,
                      (__bf16)0.f,(__bf16)0.f,(__bf16)0.f,(__bf16)0.f};
            *(v8bf*)(&Ws[0][(srow + 32 * j) * BK + scol]) = z;
            *(v8bf*)(&Ws[1][(srow + 32 * j) * BK + scol]) = z;
        }
    }

    // ---- fragment LDS offsets (elements), per ISA 7.12.2 16-bit layout ----
    int afo[4], bfo[4];
#pragma unroll
    for (int mi = 0; mi < 4; ++mi) afo[mi] = (wm * 64 + mi * 16 + l15) * BK + half * 8;
#pragma unroll
    for (int ni = 0; ni < 4; ++ni) bfo[ni] = (wn * 64 + ni * 16 + l15) * BK + half * 8;

    v8f acc[4][4];
#pragma unroll
    for (int mi = 0; mi < 4; ++mi)
#pragma unroll
        for (int ni = 0; ni < 4; ++ni)
            acc[mi][ni] = (v8f){0.f,0.f,0.f,0.f,0.f,0.f,0.f,0.f};

    // ---- prologue: async-stage tile k0=0 into buffer 0 ----
#pragma unroll
    for (int j = 0; j < 4; ++j)
        async_ld_b128(aL + j * JSTRIDE, A, aoff[j]);
#pragma unroll
    for (int j = 0; j < 8; ++j)
        if (wvalid[j]) async_ld_b128(wL + j * JSTRIDE, W, woff[j]);
    wait_async0();
    __syncthreads();

    int cur = 0;
    for (int k0 = 0; k0 < K; k0 += BK) {
        // ---- prefetch next tile into the other buffer (no wait yet) ----
        const int kn = k0 + BK;
        if (kn < K) {
            const int kb  = kn * 2;
            const unsigned ab = aL + (cur ^ 1) * ABUF;
            const unsigned wb = wL + (cur ^ 1) * WBUF;
#pragma unroll
            for (int j = 0; j < 4; ++j)
                async_ld_b128(ab + j * JSTRIDE, A, aoff[j] + kb);
#pragma unroll
            for (int j = 0; j < 8; ++j)
                if (wvalid[j]) async_ld_b128(wb + j * JSTRIDE, W, woff[j] + kb);
        }

        // ---- 2 k-steps x 16 WMMA on current buffer ----
        const __bf16* Ab = &As[cur][0];
        const __bf16* Wb = &Ws[cur][0];
#pragma unroll
        for (int ks = 0; ks < 2; ++ks) {
            v16bf bfrag[4];
#pragma unroll
            for (int ni = 0; ni < 4; ++ni) {
                const __bf16* p = Wb + bfo[ni] + ks * 32;
                v8bf lo = *(const v8bf*)p;
                v8bf hi = *(const v8bf*)(p + 16);
                bfrag[ni] = __builtin_shufflevector(lo, hi,
                    0,1,2,3,4,5,6,7,8,9,10,11,12,13,14,15);
            }
#pragma unroll
            for (int mi = 0; mi < 4; ++mi) {
                const __bf16* p = Ab + afo[mi] + ks * 32;
                v8bf lo = *(const v8bf*)p;
                v8bf hi = *(const v8bf*)(p + 16);
                v16bf afrag = __builtin_shufflevector(lo, hi,
                    0,1,2,3,4,5,6,7,8,9,10,11,12,13,14,15);
#pragma unroll
                for (int ni = 0; ni < 4; ++ni)
                    acc[mi][ni] = __builtin_amdgcn_wmma_f32_16x16x32_bf16(
                        false, afrag, false, bfrag[ni],
                        (short)0, acc[mi][ni], false, false);
            }
        }
        wait_async0();     // next buffer's asyncs landed
        __syncthreads();   // visible to all waves; current buffer free
        cur ^= 1;
    }

    // ---- epilogue: VGPR r -> M = m0 + half*8 + r ; N = n0 + l15 ----
#pragma unroll
    for (int mi = 0; mi < 4; ++mi) {
#pragma unroll
        for (int ni = 0; ni < 4; ++ni) {
            int n = n_block + wn * 64 + ni * 16 + l15;
            if (n < N) {
#pragma unroll
                for (int r = 0; r < 8; ++r) {
                    int m = m_block + wm * 64 + mi * 16 + half * 8 + r;
                    float v = acc[mi][ni][r];
                    if (mode == 1) {
                        v += bias[n];
                        v = (v > 20.f) ? v : log1pf(__expf(v));  // softplus
                    }
                    C[(size_t)m * ldc + n] = v;
                    if (Cbf) Cbf[(size_t)m * ldc + n] = (__bf16)v;
                }
            }
        }
    }
}

// ---------------------------------------------------------------------------
// One-shot f32 -> bf16 conversion (8 elements per thread).
// ---------------------------------------------------------------------------
__global__ __launch_bounds__(256) void cvt_f32_bf16(
    const float* __restrict__ src, __bf16* __restrict__ dst, int n8)
{
    int i = blockIdx.x * 256 + threadIdx.x;
    if (i < n8) {
        float4 a = ((const float4*)src)[2 * i];
        float4 b = ((const float4*)src)[2 * i + 1];
        v8bf o = {(__bf16)a.x, (__bf16)a.y, (__bf16)a.z, (__bf16)a.w,
                  (__bf16)b.x, (__bf16)b.y, (__bf16)b.z, (__bf16)b.w};
        ((v8bf*)dst)[i] = o;
    }
}

// ---------------------------------------------------------------------------
// Causal depthwise conv (K=4) + SiLU; emits f32 (for scan) + bf16 (for GEMM).
// ---------------------------------------------------------------------------
__global__ __launch_bounds__(256) void conv_silu_kernel(
    const float* __restrict__ xz,     // (B*L, 8192); x = cols [0,4096)
    const float* __restrict__ conv_w, // (4096, 4)
    const float* __restrict__ conv_b, // (4096)
    float* __restrict__ xconv,        // (B*L, 4096) f32
    __bf16* __restrict__ xconv_bf)    // (B*L, 4096) bf16
{
    int idx = blockIdx.x * 256 + threadIdx.x;   // over B*L*4096
    int d  = idx & 4095;
    int bl = idx >> 12;
    int b  = bl >> 10;
    int l  = bl & 1023;
    float4 w = *(const float4*)(conv_w + d * 4);
    float wv[4] = {w.x, w.y, w.z, w.w};
    float acc = conv_b[d];
    const float* base = xz + (size_t)b * 1024 * 8192 + d;
#pragma unroll
    for (int k = 0; k < 4; ++k) {
        int ls = l - 3 + k;
        if (ls >= 0) acc += base[(size_t)ls * 8192] * wv[k];
    }
    float s = acc / (1.f + __expf(-acc));       // SiLU
    xconv[(size_t)bl * 4096 + d]    = s;
    xconv_bf[(size_t)bl * 4096 + d] = (__bf16)s;
}

// ---------------------------------------------------------------------------
// Selective scan: one (batch, channel) per lane; 16-state recurrence in VGPRs.
// B/C vectors staged through LDS in chunks of 8 timesteps.
// Fused (y + x*D) * silu(z) gate; output bf16 (feeds out_proj WMMA).
// ---------------------------------------------------------------------------
#define LCHUNK 8
__global__ __launch_bounds__(256) void scan_kernel(
    const float* __restrict__ delta,  // (B*L, 4096) post-softplus
    const float* __restrict__ x_dbl,  // (B*L, 160): [128,144)=B, [144,160)=C
    const float* __restrict__ xconv,  // (B*L, 4096)
    const float* __restrict__ xz,     // (B*L, 8192); z at col 4096+d
    const float* __restrict__ Amat,   // (4096, 16)
    const float* __restrict__ Dp,     // (4096)
    __bf16* __restrict__ y)           // (B*L, 4096) bf16
{
    __shared__ float sBC[LCHUNK * 32];
    int tid = threadIdx.x;
    int ch  = blockIdx.x * 256 + tid;  // 0..8191; b uniform per block
    int b   = ch >> 12;
    int d   = ch & 4095;

    float Ar[16], state[16];
#pragma unroll
    for (int n = 0; n < 16; ++n) { Ar[n] = Amat[d * 16 + n]; state[n] = 0.f; }
    float Dd = Dp[d];

    for (int t0 = 0; t0 < 1024; t0 += LCHUNK) {
        __syncthreads();
        if (tid < LCHUNK * 32) {
            int tt = tid >> 5;
            int e  = tid & 31;   // 0-15 -> B[n], 16-31 -> C[n]
            sBC[tid] = x_dbl[(size_t)(b * 1024 + t0 + tt) * 160 + 128 + e];
        }
        __syncthreads();
#pragma unroll
        for (int tt = 0; tt < LCHUNK; ++tt) {
            size_t rbl = (size_t)(b * 1024 + t0 + tt);
            float dt = delta[rbl * 4096 + d];
            float xt = xconv[rbl * 4096 + d];
            float zt = xz[rbl * 8192 + 4096 + d];
            const float* Bc = sBC + tt * 32;
            float yt = 0.f;
#pragma unroll
            for (int n = 0; n < 16; ++n) {
                float dA = __expf(dt * Ar[n]);
                float s  = dA * state[n] + dt * Bc[n] * xt;
                state[n] = s;
                yt += s * Bc[16 + n];
            }
            float g = zt / (1.f + __expf(-zt));
            y[rbl * 4096 + d] = (__bf16)((yt + xt * Dd) * g);
        }
    }
}

// ---------------------------------------------------------------------------
extern "C" void kernel_launch(void* const* d_in, const int* in_sizes, int n_in,
                              void* d_out, int out_size, void* d_ws, size_t ws_size,
                              hipStream_t stream) {
    const float* hidden     = (const float*)d_in[0];
    const float* in_proj_w  = (const float*)d_in[1];
    const float* conv_w     = (const float*)d_in[2];
    const float* conv_b     = (const float*)d_in[3];
    const float* x_proj_w   = (const float*)d_in[4];
    const float* dt_proj_w  = (const float*)d_in[5];
    const float* dt_proj_b  = (const float*)d_in[6];
    const float* out_proj_w = (const float*)d_in[7];
    const float* Amat       = (const float*)d_in[8];
    const float* Dparam     = (const float*)d_in[9];
    float* out = (float*)d_out;

    // ---- workspace carve-up ----
    float* wsf   = (float*)d_ws;
    float* xz    = wsf;                          // 2048*8192 f32
    float* xconv = xz    + (size_t)2048 * 8192;  // 2048*4096 f32
    float* xdbl  = xconv + (size_t)2048 * 4096;  // 2048*160  f32
    float* delta = xdbl  + (size_t)2048 * 160;   // 2048*4096 f32
    __bf16* bws  = (__bf16*)(delta + (size_t)2048 * 4096);
    __bf16* hid_bf  = bws;                              // 2048*2048
    __bf16* inw_bf  = hid_bf  + (size_t)2048 * 2048;    // 8192*2048
    __bf16* xpw_bf  = inw_bf  + (size_t)8192 * 2048;    // 160*4096
    __bf16* dtw_bf  = xpw_bf  + (size_t)160  * 4096;    // 4096*128
    __bf16* outw_bf = dtw_bf  + (size_t)4096 * 128;     // 2048*4096
    __bf16* xconv_bf= outw_bf + (size_t)2048 * 4096;    // 2048*4096
    __bf16* xdbl_bf = xconv_bf+ (size_t)2048 * 4096;    // 2048*160
    __bf16* ybf     = xdbl_bf + (size_t)2048 * 160;     // 2048*4096

    dim3 blk(256);
    auto cvt = [&](const float* s, __bf16* d, size_t n) {
        int n8 = (int)(n / 8);
        cvt_f32_bf16<<<dim3((n8 + 255) / 256), blk, 0, stream>>>(s, d, n8);
    };

    // 0) one-shot bf16 conversions (weights + hidden)
    cvt(hidden,     hid_bf,  (size_t)2048 * 2048);
    cvt(in_proj_w,  inw_bf,  (size_t)8192 * 2048);
    cvt(x_proj_w,   xpw_bf,  (size_t)160  * 4096);
    cvt(dt_proj_w,  dtw_bf,  (size_t)4096 * 128);
    cvt(out_proj_w, outw_bf, (size_t)2048 * 4096);

    // 1) in_proj: (2048x2048) x (8192x2048)^T -> xz (2048x8192) f32
    gemm_bf16_async<<<dim3(8192 / BN, 2048 / BM), blk, 0, stream>>>(
        hid_bf, 2048, inw_bf, 2048, xz, 8192, nullptr,
        2048, 8192, 2048, 0, nullptr);

    // 2) causal depthwise conv + SiLU
    conv_silu_kernel<<<dim3((2048 * 4096) / 256), blk, 0, stream>>>(
        xz, conv_w, conv_b, xconv, xconv_bf);

    // 3) x_proj: (2048x4096) x (160x4096)^T -> x_dbl f32 + bf16
    gemm_bf16_async<<<dim3(1, 2048 / BM), blk, 0, stream>>>(
        xconv_bf, 4096, xpw_bf, 4096, xdbl, 160, xdbl_bf,
        2048, 160, 4096, 0, nullptr);

    // 4) dt_proj + softplus(.+bias): (2048x128) x (4096x128)^T -> delta f32
    gemm_bf16_async<<<dim3(4096 / BN, 2048 / BM), blk, 0, stream>>>(
        xdbl_bf, 160, dtw_bf, 128, delta, 4096, nullptr,
        2048, 4096, 128, 1, dt_proj_b);

    // 5) selective scan + gating -> y bf16
    scan_kernel<<<dim3(32), blk, 0, stream>>>(
        delta, xdbl, xconv, xz, Amat, Dparam, ybf);

    // 6) out_proj: (2048x4096) x (2048x4096)^T -> out (2048x2048) f32
    gemm_bf16_async<<<dim3(2048 / BN, 2048 / BM), blk, 0, stream>>>(
        ybf, 4096, outw_bf, 4096, out, 2048, nullptr,
        2048, 2048, 4096, 0, nullptr);
}
